// MambaBlock_8392366097156
// MI455X (gfx1250) — matrix operand order repaired and checked by reading, not compiled, not run on hardware
//
#include <hip/hip_runtime.h>
#include <hip/hip_bf16.h>
#include <math.h>

typedef _Float16 half_t;
typedef __attribute__((ext_vector_type(16))) _Float16 v16h;
typedef __attribute__((ext_vector_type(8)))  float    v8f;
typedef int v4i_vs __attribute__((vector_size(4 * sizeof(int))));

#define AS_GLOBAL __attribute__((address_space(1)))
#define AS_SHARED __attribute__((address_space(3)))

#define BD 8
#define TT 4096
#define DD 1024
#define HH 64
#define MM (BD*TT)   // 32768 token rows

// ---- CDNA5 async VMEM->LDS path (guarded: falls back to reg staging) ------
#if __has_builtin(__builtin_amdgcn_global_load_async_to_lds_b128)
#define GEMM_ASYNC_COPY 1
#else
#define GEMM_ASYNC_COPY 0
#endif

#if __has_builtin(__builtin_amdgcn_s_wait_asynccnt)
#define WAIT_ASYNC0() __builtin_amdgcn_s_wait_asynccnt(0)
#else
#define WAIT_ASYNC0() asm volatile("s_wait_asynccnt 0x0" ::: "memory")
#endif

// ---------------------------------------------------------------------------
// f32 -> f16 conversion (weights)
// ---------------------------------------------------------------------------
__global__ void cvt_f16_kernel(const float* __restrict__ src,
                               half_t* __restrict__ dst, int n) {
  int i = blockIdx.x * 256 + threadIdx.x;
  if (i < n) dst[i] = (half_t)src[i];
}

// pack [db|Bb] bias and A_log = -softplus(A)
__global__ void pack_small_kernel(const float* __restrict__ db,
                                  const float* __restrict__ Bb,
                                  const float* __restrict__ A,
                                  float* __restrict__ bias128,
                                  float* __restrict__ Alog) {
  int i = threadIdx.x;
  if (i < HH) {
    bias128[i]      = db[i];
    bias128[HH + i] = Bb[i];
    float a = A[i];
    Alog[i] = -(log1pf(expf(-fabsf(a))) + fmaxf(a, 0.f)); // -softplus
  }
}

// ---------------------------------------------------------------------------
// RMSNorm: one wave32 per row of D=1024, output f16
// ---------------------------------------------------------------------------
__global__ __launch_bounds__(256)
void rmsnorm_kernel(const float* __restrict__ x, const float* __restrict__ w,
                    half_t* __restrict__ xn) {
  int row  = blockIdx.x * 8 + (threadIdx.x >> 5);
  int lane = threadIdx.x & 31;
  const float* xr = x + (size_t)row * DD;
  float4 v[8];
  float s = 0.f;
#pragma unroll
  for (int j = 0; j < 8; ++j) {
    v[j] = ((const float4*)xr)[lane + 32 * j];
    s += v[j].x * v[j].x + v[j].y * v[j].y + v[j].z * v[j].z + v[j].w * v[j].w;
  }
#pragma unroll
  for (int off = 16; off > 0; off >>= 1) s += __shfl_xor(s, off, 32);
  float r = rsqrtf(s * (1.f / DD) + 1.1920928955078125e-7f);
  half_t* o = xn + (size_t)row * DD;
#pragma unroll
  for (int j = 0; j < 8; ++j) {
    int c = (lane + 32 * j) * 4;
    o[c + 0] = (half_t)(v[j].x * r * w[c + 0]);
    o[c + 1] = (half_t)(v[j].y * r * w[c + 1]);
    o[c + 2] = (half_t)(v[j].z * r * w[c + 2]);
    o[c + 3] = (half_t)(v[j].w * r * w[c + 3]);
  }
}

// ---------------------------------------------------------------------------
// WMMA GEMM: C[M,N] = A[M,K] * W[N,K]^T (+bias +accumulate +residual)
// block tile 128x128, 8 waves, wave tile 32x64, K chunk 32, double-buffered
// LDS with async VMEM->LDS staging (ASYNCcnt). Epilogue variants templated.
// ---------------------------------------------------------------------------
#define TILE_M 128
#define TILE_N 128
#define TILE_K 32
#define LDSW   40   // halves per row (80B, keeps 16B alignment for b128 copies)

// 16-bit A-matrix 16x32 fragment (ISA 7.12.2): lanes 0-15 -> M rows, K split
// {0..7,16..23} / lanes 16-31 -> K {8..15,24..31}, 2 halves per VGPR.
__device__ inline v16h load_a_frag(const half_t* s, int rowBase, int lane) {
  const half_t* p = s + (rowBase + (lane & 15)) * LDSW;
  int kb = (lane >> 4) * 8;
  v16h v;
#pragma unroll
  for (int j = 0; j < 8; ++j) {
    int k = ((j < 4) ? 0 : 16) + kb + 2 * (j & 3);
    v[2 * j]     = p[k];
    v[2 * j + 1] = p[k + 1];
  }
  return v;
}

// 16-bit B-matrix 32x16 fragment: lanes 0-15 -> K {0..7,16..23}, lanes 16-31 -> K {8..15,24..31},
// column = lane&15; LDS tile holds W rows (N-major, K contiguous).
__device__ inline v16h load_b_frag(const half_t* s, int colBase, int lane) {
  const half_t* p = s + (colBase + (lane & 15)) * LDSW;
  int kb = (lane >> 4) * 8;
  v16h v;
#pragma unroll
  for (int j = 0; j < 8; ++j) {
    int k = ((j < 4) ? 0 : 16) + kb + 2 * (j & 3);
    v[2 * j]     = p[k];
    v[2 * j + 1] = p[k + 1];
  }
  return v;
}

#if GEMM_ASYNC_COPY
// per-thread 32B (2 x b128) async copy; IOFFSET applies to both sides
__device__ inline void async_cp32(const half_t* g, half_t* s) {
  __builtin_amdgcn_global_load_async_to_lds_b128(
      (AS_GLOBAL v4i_vs*)g, (AS_SHARED v4i_vs*)s, 0, 0);
  __builtin_amdgcn_global_load_async_to_lds_b128(
      (AS_GLOBAL v4i_vs*)g, (AS_SHARED v4i_vs*)s, 16, 0);
}
#endif

template <bool HAS_BIAS, bool ACCUM, bool HAS_RESID>
__global__ __launch_bounds__(256, 2)
void wmma_gemm_kernel(const half_t* __restrict__ A, const half_t* __restrict__ W,
                      float* __restrict__ C, const float* __restrict__ bias,
                      const float* __restrict__ resid, int N, int K) {
  __shared__ half_t As[2][TILE_M * LDSW];
  __shared__ half_t Bs[2][TILE_N * LDSW];
  int tid  = threadIdx.x;
  int lane = tid & 31;
  int wave = tid >> 5;
  int wm = (wave >> 1) * 32;   // wave row offset: 0,32,64,96
  int wn = (wave & 1) * 64;    // wave col offset: 0,64
  int mBase = blockIdx.x * TILE_M;
  int nBase = blockIdx.y * TILE_N;

  v8f acc[2][4];
#pragma unroll
  for (int i = 0; i < 2; ++i)
#pragma unroll
    for (int j = 0; j < 4; ++j)
#pragma unroll
      for (int r = 0; r < 8; ++r) acc[i][j][r] = 0.f;

  int kChunks = K / TILE_K;
  int lr = tid >> 1;          // tile row 0..127
  int lc = (tid & 1) * 16;    // half-offset 0 or 16 (32B)
  const half_t* gA = A + (size_t)(mBase + lr) * K + lc;
  const half_t* gB = W + (size_t)(nBase + lr) * K + lc;
  int sOff = lr * LDSW + lc;

  // ---- prologue: stage chunk 0 into buffer 0 ----
#if GEMM_ASYNC_COPY
  async_cp32(gA, &As[0][sOff]);
  async_cp32(gB, &Bs[0][sOff]);
  WAIT_ASYNC0();
#else
  {
    int4 a0 = ((const int4*)gA)[0];
    int4 a1 = ((const int4*)gA)[1];
    int4 b0 = ((const int4*)gB)[0];
    int4 b1 = ((const int4*)gB)[1];
    ((int4*)&As[0][sOff])[0] = a0; ((int4*)&As[0][sOff])[1] = a1;
    ((int4*)&Bs[0][sOff])[0] = b0; ((int4*)&Bs[0][sOff])[1] = b1;
  }
#endif
  __syncthreads();
  gA += TILE_K; gB += TILE_K;

  for (int kc = 0; kc < kChunks; ++kc) {
    int cur = kc & 1;
    int nxt = cur ^ 1;
    bool more = (kc + 1 < kChunks);

    // ---- start staging next chunk (overlaps with WMMA below) ----
#if GEMM_ASYNC_COPY
    if (more) {
      async_cp32(gA, &As[nxt][sOff]);
      async_cp32(gB, &Bs[nxt][sOff]);
      gA += TILE_K; gB += TILE_K;
    }
#else
    int4 a0, a1, b0, b1;
    if (more) {
      a0 = ((const int4*)gA)[0];
      a1 = ((const int4*)gA)[1];
      b0 = ((const int4*)gB)[0];
      b1 = ((const int4*)gB)[1];
      gA += TILE_K; gB += TILE_K;
      if (kc + 2 < kChunks) {      // global_prefetch_b8 for the chunk after
        __builtin_prefetch(gA, 0, 0);
        __builtin_prefetch(gB, 0, 0);
      }
    }
#endif

    // ---- compute current chunk ----
    v16h af[2], bf[4];
#pragma unroll
    for (int i = 0; i < 2; ++i) af[i] = load_a_frag(As[cur], wm + i * 16, lane);
#pragma unroll
    for (int j = 0; j < 4; ++j) bf[j] = load_b_frag(Bs[cur], wn + j * 16, lane);
#pragma unroll
    for (int i = 0; i < 2; ++i)
#pragma unroll
      for (int j = 0; j < 4; ++j)
        acc[i][j] = __builtin_amdgcn_wmma_f32_16x16x32_f16(
            false, af[i], false, bf[j], (short)0, acc[i][j], false, false);

    // ---- finish staging, publish next buffer ----
#if GEMM_ASYNC_COPY
    if (more) WAIT_ASYNC0();
#else
    if (more) {
      ((int4*)&As[nxt][sOff])[0] = a0; ((int4*)&As[nxt][sOff])[1] = a1;
      ((int4*)&Bs[nxt][sOff])[0] = b0; ((int4*)&Bs[nxt][sOff])[1] = b1;
    }
#endif
    __syncthreads();
  }

  // epilogue (straight-line per template variant):
  // C/D layout VGPR r -> M=(lane>=16?8:0)+r (+16*i), N=lane&15
  int rowHalf = (lane >> 4) * 8;
  int colIn   = lane & 15;
#pragma unroll
  for (int i = 0; i < 2; ++i) {
#pragma unroll
    for (int j = 0; j < 4; ++j) {
      int col = nBase + wn + j * 16 + colIn;
      float bval = 0.f;
      if (HAS_BIAS) bval = bias[col];
      size_t base = (size_t)(mBase + wm + i * 16 + rowHalf) * N + col;

      float oldv[8], resv[8];
      if (ACCUM) {
#pragma unroll
        for (int r = 0; r < 8; ++r) oldv[r] = C[base + (size_t)r * N];
      }
      if (HAS_RESID) {
#pragma unroll
        for (int r = 0; r < 8; ++r) resv[r] = resid[base + (size_t)r * N];
      }
#pragma unroll
      for (int r = 0; r < 8; ++r) {
        float v = acc[i][j][r];
        if (HAS_BIAS)  v += bval;
        if (ACCUM)     v += oldv[r];
        if (HAS_RESID) v += resv[r];
        C[base + (size_t)r * N] = v;
      }
    }
  }
}

// ---------------------------------------------------------------------------
// causal depthwise conv(k=3) + SiLU; reads x_b = xbz[:, 0:1024], writes f16 xc
// ---------------------------------------------------------------------------
__global__ void conv_silu_kernel(const float* __restrict__ xbz,
                                 const float* __restrict__ cw,
                                 const float* __restrict__ cb,
                                 half_t* __restrict__ xc) {
  int i = blockIdx.x * 256 + threadIdx.x;
  if (i >= MM * DD) return;
  int d = i & (DD - 1);
  int m = i >> 10;
  int t = m & (TT - 1);
  const float* col = xbz + (size_t)m * 2048 + d;
  float v = col[0] * cw[d * 3 + 2];
  if (t >= 1) v += col[-2048]   * cw[d * 3 + 1];
  if (t >= 2) v += col[-2*2048] * cw[d * 3 + 0];
  v += cb[d];
  v = v / (1.f + expf(-v));    // SiLU
  xc[i] = (half_t)v;
}

// ---------------------------------------------------------------------------
// delta/B_bar prep: dBv[M,128] = [xc@dw^T + db | xc@Bw^T + Bb]
// a = softplus(.)*A_log[h], bb = softplus(.)*(.);  output layout (b*H+h, T)
// ---------------------------------------------------------------------------
__global__ void scanprep_kernel(const float* __restrict__ dBv,
                                const float* __restrict__ Alog,
                                float* __restrict__ a_buf,
                                float* __restrict__ bb_buf) {
  int i = blockIdx.x * 256 + threadIdx.x;
  if (i >= MM * HH) return;
  int h = i & (HH - 1);
  int m = i >> 6;
  int b = m >> 12;
  int t = m & (TT - 1);
  float raw   = dBv[(size_t)m * 128 + h];
  float delta = log1pf(expf(-fabsf(raw))) + fmaxf(raw, 0.f);  // softplus
  size_t o = ((size_t)(b * HH + h)) * TT + t;
  a_buf[o]  = delta * Alog[h];
  bb_buf[o] = delta * dBv[(size_t)m * 128 + HH + h];
}

// ---------------------------------------------------------------------------
// selective scan: one block per (b,h); chunk=16/thread; two block scans
// (add-scan for clipped cumsum carry, operator scan for the recurrence)
// ---------------------------------------------------------------------------
__global__ __launch_bounds__(256)
void scan_kernel(const float* __restrict__ a_buf,
                 const float* __restrict__ bb_buf,
                 half_t* __restrict__ hbuf) {
  __shared__ float sE[256];
  __shared__ float sB[256];
  int seq = blockIdx.x;          // b*64 + h
  int b   = seq >> 6;
  int hh  = seq & 63;
  int tid = threadIdx.x;

  const float* ap = a_buf  + (size_t)seq * TT + tid * 16;
  const float* bp = bb_buf + (size_t)seq * TT + tid * 16;
  float a[16], bb[16];
#pragma unroll
  for (int j = 0; j < 16; ++j) { a[j] = ap[j]; bb[j] = bp[j]; }

  float suma = 0.f;
#pragma unroll
  for (int j = 0; j < 16; ++j) suma += a[j];

  // block inclusive add-scan (Hillis-Steele) -> exclusive carry
  sE[tid] = suma;
  __syncthreads();
  for (int off = 1; off < 256; off <<= 1) {
    float v = sE[tid];
    float o = (tid >= off) ? sE[tid - off] : 0.f;
    __syncthreads();
    sE[tid] = v + o;
    __syncthreads();
  }
  float carry = sE[tid] - suma;   // unclipped cumsum entering this chunk
  __syncthreads();

  // per-step decay f_j from clipped cumsum; chunk operator (E, Bc)
  float gprev = fminf(fmaxf(carry, -20.f), 20.f);
  float run = carry;
  float f[16];
  float E = 1.f, Bc = 0.f;
#pragma unroll
  for (int j = 0; j < 16; ++j) {
    run += a[j];
    float g = fminf(fmaxf(run, -20.f), 20.f);
    float fj = expf(g - gprev);
    gprev = g;
    f[j] = fj;
    E  = fj * E;
    Bc = fj * (Bc + bb[j]);
  }

  // block operator scan: (E1,B1) then (E2,B2) => (E2*E1, E2*B1 + B2)
  sE[tid] = E; sB[tid] = Bc;
  __syncthreads();
  for (int off = 1; off < 256; off <<= 1) {
    float e2 = sE[tid], b2 = sB[tid];
    float e1 = (tid >= off) ? sE[tid - off] : 1.f;
    float b1 = (tid >= off) ? sB[tid - off] : 0.f;
    __syncthreads();
    sE[tid] = e2 * e1;
    sB[tid] = e2 * b1 + b2;
    __syncthreads();
  }
  float hin = (tid == 0) ? 0.f : sB[tid - 1];   // state at chunk entry (s0=0)

  int tbase = tid * 16;
  float hcur = hin;
#pragma unroll
  for (int j = 0; j < 16; ++j) {
    hcur = f[j] * (hcur + bb[j]);
    size_t o = (((size_t)b * TT) + (tbase + j)) * HH + hh;
    hbuf[o] = (half_t)hcur;
  }
}

// ---------------------------------------------------------------------------
// gating: g = y * silu(z), z = xbz[:, 1024:2048]; output f16
// ---------------------------------------------------------------------------
__global__ void gate_kernel(const float* __restrict__ y,
                            const float* __restrict__ xbz,
                            half_t* __restrict__ g) {
  int i = blockIdx.x * 256 + threadIdx.x;
  if (i >= MM * DD) return;
  int d = i & (DD - 1);
  size_t m = (size_t)(i >> 10);
  float z = xbz[m * 2048 + 1024 + d];
  float sv = z / (1.f + expf(-z));
  g[i] = (half_t)(y[i] * sv);
}

// ---------------------------------------------------------------------------
extern "C" void kernel_launch(void* const* d_in, const int* in_sizes, int n_in,
                              void* d_out, int out_size, void* d_ws, size_t ws_size,
                              hipStream_t stream) {
  const float* x      = (const float*)d_in[0];
  const float* norm_w = (const float*)d_in[1];
  const float* W1     = (const float*)d_in[2];
  const float* W2     = (const float*)d_in[3];
  const float* Wlast  = (const float*)d_in[4];
  const float* conv_w = (const float*)d_in[5];
  const float* conv_b = (const float*)d_in[6];
  const float* A      = (const float*)d_in[7];
  const float* Bw     = (const float*)d_in[8];
  const float* Bb     = (const float*)d_in[9];
  const float* Cw     = (const float*)d_in[10];
  const float* Cb     = (const float*)d_in[11];
  const float* Dw     = (const float*)d_in[12];
  const float* Db     = (const float*)d_in[13];
  const float* dw     = (const float*)d_in[14];
  const float* db     = (const float*)d_in[15];
  float* out = (float*)d_out;

  char* ws = (char*)d_ws;
  size_t off = 0;
  auto alloc = [&](size_t bytes) {
    size_t o = off;
    off += (bytes + 255) & ~(size_t)255;
    return o;
  };
  half_t* xn      = (half_t*)(ws + alloc((size_t)MM * DD * 2));      // also reused as g
  half_t* w12     = (half_t*)(ws + alloc((size_t)2048 * 1024 * 2));
  half_t* wdB     = (half_t*)(ws + alloc((size_t)128 * 1024 * 2));
  half_t* wD      = (half_t*)(ws + alloc((size_t)1024 * 1024 * 2));
  half_t* wC      = (half_t*)(ws + alloc((size_t)1024 * 64 * 2));
  half_t* wL      = (half_t*)(ws + alloc((size_t)1024 * 1024 * 2));
  float*  bias128 = (float*)(ws + alloc(128 * 4));
  float*  Alog    = (float*)(ws + alloc(64 * 4));
  float*  xbz     = (float*)(ws + alloc((size_t)MM * 2048 * 4));
  half_t* xc      = (half_t*)(ws + alloc((size_t)MM * DD * 2));
  float*  dBv     = (float*)(ws + alloc((size_t)MM * 128 * 4));
  float*  a_buf   = (float*)(ws + alloc((size_t)MM * HH * 4));
  float*  bb_buf  = (float*)(ws + alloc((size_t)MM * HH * 4));
  half_t* hbuf    = (half_t*)(ws + alloc((size_t)MM * HH * 2));
  float*  ybuf    = (float*)(ws + alloc((size_t)MM * DD * 4));

  const int DxD = 1024 * 1024;
  cvt_f16_kernel<<<(DxD + 255) / 256, 256, 0, stream>>>(W1, w12, DxD);
  cvt_f16_kernel<<<(DxD + 255) / 256, 256, 0, stream>>>(W2, w12 + DxD, DxD);
  cvt_f16_kernel<<<(64 * 1024 + 255) / 256, 256, 0, stream>>>(dw, wdB, 64 * 1024);
  cvt_f16_kernel<<<(64 * 1024 + 255) / 256, 256, 0, stream>>>(Bw, wdB + 64 * 1024, 64 * 1024);
  cvt_f16_kernel<<<(DxD + 255) / 256, 256, 0, stream>>>(Dw, wD, DxD);
  cvt_f16_kernel<<<(1024 * 64 + 255) / 256, 256, 0, stream>>>(Cw, wC, 1024 * 64);
  cvt_f16_kernel<<<(DxD + 255) / 256, 256, 0, stream>>>(Wlast, wL, DxD);
  pack_small_kernel<<<1, 64, 0, stream>>>(db, Bb, A, bias128, Alog);

  rmsnorm_kernel<<<MM / 8, 256, 0, stream>>>(x, norm_w, xn);

  // xbz = xn @ [W1|W2]^T   (M x 2048)
  dim3 g1(MM / TILE_M, 2048 / TILE_N);
  wmma_gemm_kernel<false, false, false>
      <<<g1, 256, 0, stream>>>(xn, w12, xbz, nullptr, nullptr, 2048, 1024);

  conv_silu_kernel<<<(MM * DD + 255) / 256, 256, 0, stream>>>(xbz, conv_w, conv_b, xc);

  // dBv = xc @ [dw|Bw]^T + [db|Bb]   (M x 128)
  dim3 g2(MM / TILE_M, 1);
  wmma_gemm_kernel<true, false, false>
      <<<g2, 256, 0, stream>>>(xc, wdB, dBv, bias128, nullptr, 128, 1024);

  scanprep_kernel<<<(MM * HH + 255) / 256, 256, 0, stream>>>(dBv, Alog, a_buf, bb_buf);
  scan_kernel<<<BD * HH, 256, 0, stream>>>(a_buf, bb_buf, hbuf);

  // y = xc @ Dw^T + Db  ;  y += h @ Cw^T + Cb
  dim3 g3(MM / TILE_M, 1024 / TILE_N);
  wmma_gemm_kernel<true, false, false>
      <<<g3, 256, 0, stream>>>(xc, wD, ybuf, Db, nullptr, 1024, 1024);
  wmma_gemm_kernel<true, true, false>
      <<<g3, 256, 0, stream>>>(hbuf, wC, ybuf, Cb, nullptr, 1024, 64);

  // g = y * silu(z)  (reuse xn buffer)
  gate_kernel<<<(MM * DD + 255) / 256, 256, 0, stream>>>(ybuf, xbz, xn);

  // out = g @ Wlast^T + x
  wmma_gemm_kernel<false, false, true>
      <<<g3, 256, 0, stream>>>(xn, wL, out, nullptr, x, 1024, 1024);
}
